// SplashAttention_37357625541119
// MI455X (gfx1250) — matrix operand order repaired and checked by reading, not compiled, not run on hardware
//
#include <hip/hip_runtime.h>

#define B_ 2
#define H_ 16
#define S_ 2048
#define D_ 128
#define QTILE 64
#define KTILE 32
#define NEG_INF (-1.0e30f)

typedef __attribute__((ext_vector_type(16))) __bf16 v16bf;
typedef __attribute__((ext_vector_type(8)))  __bf16 v8bf;
typedef __attribute__((ext_vector_type(8)))  float  v8f;

__global__ __launch_bounds__(128) void splash_fwd(
    const float* __restrict__ Qg, const float* __restrict__ Kg,
    const float* __restrict__ Vg, const int* __restrict__ qsegg,
    const int* __restrict__ ksegg, const int* __restrict__ winp,
    float* __restrict__ Og)
{
    // K tile row-major bf16, V tile transposed bf16, per-wave P scratch
    __shared__ __align__(32) __bf16 Kl[KTILE][D_];      // 8 KB
    __shared__ __align__(32) __bf16 VTl[D_][KTILE];     // 8 KB
    __shared__ __align__(32) __bf16 Pl[4][16][KTILE];   // 4 KB

    const int tid  = threadIdx.x;
    const int w    = tid >> 5;      // wave id 0..3
    const int lane = tid & 31;
    const int half = lane >> 4;     // 0/1
    const int n16  = lane & 15;

    const int blk = blockIdx.x;
    const int qt  = blk & 31;       // S_/QTILE = 32
    const int bh  = blk >> 5;       // 0..31 = b*H + h
    const int b   = bh >> 4;        // H = 16

    const int win = winp[0];
    const int qb  = qt * QTILE;
    const int wq0 = qb + w * 16;    // first query row of this wave

    const float scale = 0.088388347648318447f; // 1/sqrt(128)

    // ---- Q rows in 16-bit A-operand layout, pre-scaled, bf16 ----
    // A row m = lane%16 ; lanes 0-15 hold k = c*32+[0..8)+[16..24),
    // lanes 16-31 hold k = c*32+[8..16)+[24..32).
    const float* qrow = Qg + ((size_t)bh * S_ + (wq0 + n16)) * D_;
    v16bf qa[4];
    #pragma unroll
    for (int c = 0; c < 4; ++c) {
        #pragma unroll
        for (int e = 0; e < 8; ++e) {
            qa[c][e]     = (__bf16)(qrow[c*32 + half*8 + e] * scale);
            qa[c][e + 8] = (__bf16)(qrow[c*32 + 16 + half*8 + e] * scale);
        }
    }

    // q segment ids for this lane's 8 C-layout rows (M = half*8 + r)
    int qsegr[8];
    #pragma unroll
    for (int r = 0; r < 8; ++r)
        qsegr[r] = qsegg[b * S_ + wq0 + half*8 + r];

    v8f zero = {};
    v8f o[8];
    #pragma unroll
    for (int dt = 0; dt < 8; ++dt) o[dt] = zero;
    float mrow[8], lrow[8];
    #pragma unroll
    for (int r = 0; r < 8; ++r) { mrow[r] = -__builtin_inff(); lrow[r] = 0.0f; }

    int lo = qb - win;                 if (lo < 0)  lo = 0;
    int hi = qb + (QTILE - 1) + win + 1; if (hi > S_) hi = S_;
    const int t0 = lo >> 5, t1 = (hi + 31) >> 5;

    for (int t = t0; t < t1; ++t) {
        const int base = t * 32;
        __syncthreads();   // previous tile's readers done before restage

        // ---- cooperative staging: 32 kv rows x 128 d ----
        #pragma unroll
        for (int i = 0; i < 8; ++i) {
            int idx = tid + i * 128;          // 1024 float4 groups
            int row = idx >> 5;               // kv row in tile, 0..31
            int dg  = (idx & 31) * 4;         // d group
            int kv  = base + row; if (kv > S_ - 1) kv = S_ - 1;  // masked later
            const float4 kq = *(const float4*)(Kg + ((size_t)bh * S_ + kv) * D_ + dg);
            const float4 vq = *(const float4*)(Vg + ((size_t)bh * S_ + kv) * D_ + dg);
            Kl[row][dg+0] = (__bf16)kq.x; Kl[row][dg+1] = (__bf16)kq.y;
            Kl[row][dg+2] = (__bf16)kq.z; Kl[row][dg+3] = (__bf16)kq.w;
            VTl[dg+0][row] = (__bf16)vq.x; VTl[dg+1][row] = (__bf16)vq.y;
            VTl[dg+2][row] = (__bf16)vq.z; VTl[dg+3][row] = (__bf16)vq.w;
        }
        __syncthreads();

        // wave-uniform skip if tile entirely outside this wave's window
        const bool active = !((base + 31 < wq0 - win) || (base > wq0 + 15 + win));
        if (active) {
            // ---- S = Q K^T : two 16x16 f32 tiles over 4 k-chunks ----
            v8f s0 = zero, s1 = zero;
            #pragma unroll
            for (int c = 0; c < 4; ++c) {
                // B operand: lane = kv column (n16 / 16+n16), d = c*32+half*8+[0..8) and +16
                v16bf b0 = __builtin_shufflevector(*(const v8bf*)&Kl[n16][c*32 + half*8], *(const v8bf*)&Kl[n16][c*32 + 16 + half*8], 0,1,2,3,4,5,6,7,8,9,10,11,12,13,14,15);
                v16bf b1 = __builtin_shufflevector(*(const v8bf*)&Kl[16 + n16][c*32 + half*8], *(const v8bf*)&Kl[16 + n16][c*32 + 16 + half*8], 0,1,2,3,4,5,6,7,8,9,10,11,12,13,14,15);
                s0 = __builtin_amdgcn_wmma_f32_16x16x32_bf16(
                        false, qa[c], false, b0, (short)0, s0, false, false);
                s1 = __builtin_amdgcn_wmma_f32_16x16x32_bf16(
                        false, qa[c], false, b1, (short)0, s1, false, false);
            }

            // ---- mask + online softmax (rows across 16-lane groups) ----
            const int col0 = base + n16, col1 = base + 16 + n16;
            const int ks0 = ksegg[b * S_ + (col0 < S_ ? col0 : S_ - 1)];
            const int ks1 = ksegg[b * S_ + (col1 < S_ ? col1 : S_ - 1)];
            float p0[8], p1[8], alpha[8];
            #pragma unroll
            for (int r = 0; r < 8; ++r) {
                const int qi = wq0 + half*8 + r;
                const int d0 = qi - col0, d1 = qi - col1;
                const bool v0 = (col0 < S_) && (d0 <= win) && (d0 >= -win) && (qsegr[r] == ks0);
                const bool v1 = (col1 < S_) && (d1 <= win) && (d1 >= -win) && (qsegr[r] == ks1);
                const float a0 = v0 ? s0[r] : NEG_INF;
                const float a1 = v1 ? s1[r] : NEG_INF;
                float tmax = fmaxf(a0, a1);
                tmax = fmaxf(tmax, __shfl_xor(tmax, 1));
                tmax = fmaxf(tmax, __shfl_xor(tmax, 2));
                tmax = fmaxf(tmax, __shfl_xor(tmax, 4));
                tmax = fmaxf(tmax, __shfl_xor(tmax, 8));
                const float mnew = fmaxf(mrow[r], tmax);
                alpha[r] = __expf(mrow[r] - mnew);
                p0[r] = __expf(a0 - mnew);
                p1[r] = __expf(a1 - mnew);
                float ts = p0[r] + p1[r];
                ts += __shfl_xor(ts, 1);
                ts += __shfl_xor(ts, 2);
                ts += __shfl_xor(ts, 4);
                ts += __shfl_xor(ts, 8);
                lrow[r] = lrow[r] * alpha[r] + ts;
                mrow[r] = mnew;
            }
            #pragma unroll
            for (int dt = 0; dt < 8; ++dt) {
                #pragma unroll
                for (int r = 0; r < 8; ++r) o[dt][r] *= alpha[r];
            }

            // ---- P (C layout) -> LDS -> A-operand layout, bf16 ----
            #pragma unroll
            for (int r = 0; r < 8; ++r) {
                Pl[w][half*8 + r][n16]      = (__bf16)p0[r];
                Pl[w][half*8 + r][16 + n16] = (__bf16)p1[r];
            }
            asm volatile("s_wait_dscnt 0" ::: "memory"); // wave-private RAW
            v8bf plo = *(const v8bf*)&Pl[w][n16][half*8];
            v8bf phi = *(const v8bf*)&Pl[w][n16][16 + half*8];
            v16bf pa = __builtin_shufflevector(plo, phi,
                         0,1,2,3,4,5,6,7,8,9,10,11,12,13,14,15);

            // ---- O += P @ V : 8 d-tiles of 16, K = 32 kv rows ----
            #pragma unroll
            for (int dt = 0; dt < 8; ++dt) {
                v16bf bv = __builtin_shufflevector(*(const v8bf*)&VTl[dt*16 + n16][half*8], *(const v8bf*)&VTl[dt*16 + n16][16 + half*8], 0,1,2,3,4,5,6,7,8,9,10,11,12,13,14,15);
                o[dt] = __builtin_amdgcn_wmma_f32_16x16x32_bf16(
                           false, pa, false, bv, (short)0, o[dt], false, false);
            }
        }
    }

    // ---- normalize and write out (f32) ----
    #pragma unroll
    for (int r = 0; r < 8; ++r) {
        const float inv = 1.0f / lrow[r];
        const int qi = wq0 + half*8 + r;
        float* orow = Og + ((size_t)bh * S_ + qi) * D_;
        #pragma unroll
        for (int dt = 0; dt < 8; ++dt)
            orow[dt*16 + n16] = o[dt][r] * inv;
    }
}

extern "C" void kernel_launch(void* const* d_in, const int* in_sizes, int n_in,
                              void* d_out, int out_size, void* d_ws, size_t ws_size,
                              hipStream_t stream) {
    const float* q    = (const float*)d_in[0];
    const float* k    = (const float*)d_in[1];
    const float* v    = (const float*)d_in[2];
    const int*   qseg = (const int*)d_in[3];
    const int*   kseg = (const int*)d_in[4];
    const int*   win  = (const int*)d_in[5];
    float* out = (float*)d_out;
    (void)in_sizes; (void)n_in; (void)out_size; (void)d_ws; (void)ws_size;

    dim3 grid(B_ * H_ * (S_ / QTILE));   // 1024 blocks
    dim3 block(128);                     // 4 waves of 32
    splash_fwd<<<grid, block, 0, stream>>>(q, k, v, qseg, kseg, win, out);
}
